// PETDecoder_2611340116019
// MI455X (gfx1250) — matrix operand order repaired and checked by reading, not compiled, not run on hardware
//
#include <hip/hip_runtime.h>
#include <cstdint>
#include <cstddef>

#define BB 8
#define HH 96
#define WWID 128
#define SS 12288
#define CCH 256
#define NQ 2457
#define ROWS 98304
#define RQ 19656
#define RQP 19680

typedef __attribute__((ext_vector_type(16))) _Float16 v16h;
typedef __attribute__((ext_vector_type(8))) float v8f;

// ---------------------------------------------------------------------------
// K0: valid_W / valid_H per batch from mask (vwh[0..7]=validW, vwh[8..15]=validH)
// ---------------------------------------------------------------------------
__global__ void k_valid(const unsigned char* __restrict__ mask, float* __restrict__ vwh) {
    int b = blockIdx.x, t = threadIdx.x;
    __shared__ int ch, cw;
    if (t == 0) { ch = 0; cw = 0; }
    __syncthreads();
    int lh = 0, lw = 0;
    if (t < HH) lh = mask[(size_t)b * SS + (size_t)t * WWID] ? 0 : 1;
    if (t < WWID) lw = mask[(size_t)b * SS + t] ? 0 : 1;
    atomicAdd(&ch, lh);
    atomicAdd(&cw, lw);
    __syncthreads();
    if (t == 0) { vwh[b] = (float)cw; vwh[8 + b] = (float)ch; }
}

// ---------------------------------------------------------------------------
// K1: pack 256x256 f32 weights into per-lane WMMA B-fragment layout (f16).
// Bpack[kk][nt][lane][e] = W[K][n], n = nt*16+(lane&15), K = kk*32+(lane>>4)*16+e
// ---------------------------------------------------------------------------
__global__ void k_pack(const float* __restrict__ w0, const float* __restrict__ w1,
                       const float* __restrict__ w2, const float* __restrict__ w3,
                       _Float16* __restrict__ p0, _Float16* __restrict__ p1,
                       _Float16* __restrict__ p2, _Float16* __restrict__ p3) {
    int idx = blockIdx.x * 256 + threadIdx.x;      // 0..262143
    int mat = idx >> 16;
    int o = idx & 65535;
    int e = o & 15, lane = (o >> 4) & 31, nt = (o >> 9) & 15, kk = o >> 13;
    int n = nt * 16 + (lane & 15);
    int K = kk * 32 + (lane >> 4) * 16 + e;
    const float* src = (mat == 0) ? w0 : (mat == 1) ? w1 : (mat == 2) ? w2 : w3;
    _Float16* dst = (mat == 0) ? p0 : (mat == 1) ? p1 : (mat == 2) ? p2 : p3;
    dst[o] = (_Float16)src[K * 256 + n];
}

// ---------------------------------------------------------------------------
// Core: one wave computes a 16x256 tile: C = A(16x256,f16 in LDS) * B(256x256 packed)
// 16 nt-tiles x 8 k-steps of v_wmma_f32_16x16x32_f16, f32 accum -> Clds[16][256]
// ---------------------------------------------------------------------------
__device__ __forceinline__ void wave_gemm(const _Float16* Alds,
                                          const _Float16* __restrict__ Bp,
                                          float* Clds, int lane) {
    const int m = lane & 15;
    const int kh = lane >> 4;
    v16h afr[8];
#pragma unroll
    for (int kk = 0; kk < 8; ++kk) {
        const _Float16* a0 = Alds + m * 256 + kk * 32 + kh * 16;
#pragma unroll
        for (int e = 0; e < 8; ++e) {
            afr[kk][e] = a0[e];           // K = kk*32 + kh*16 + e
            afr[kk][8 + e] = a0[8 + e];  // K = kk*32 + kh*16 + 8 + e
        }
    }
#pragma unroll
    for (int nt = 0; nt < 16; ++nt) {
        v8f acc = {0.f, 0.f, 0.f, 0.f, 0.f, 0.f, 0.f, 0.f};
#pragma unroll
        for (int kk = 0; kk < 8; ++kk) {
            v16h bfr = *reinterpret_cast<const v16h*>(Bp + (size_t)(((kk * 16) + nt) * 32 + lane) * 16);
            acc = __builtin_amdgcn_wmma_f32_16x16x32_f16(false, afr[kk], false, bfr,
                                                         (short)0, acc, false, false);
        }
#pragma unroll
        for (int j = 0; j < 8; ++j)
            Clds[(kh * 8 + j) * 256 + nt * 16 + m] = acc[j];  // D[M][N], N = nt*16 + (lane&15)
    }
}

// ---------------------------------------------------------------------------
// GEMM kernels (2 waves / block, each wave a 16-row tile).
// MODE 0: A = masked memory (f32->f16), +enc_b, LN(ng,nb) -> H0(f16); cls head -> enc_cls + scores
// MODE 1: A = H0,  relu(+b1) -> H1 (f16)
// MODE 2: A = H1,  relu(+b2) -> dot mlp_w3 (+b3 + prop_logit) -> cu_ws + flipped sigmoid coords
// MODE 3: A = posbuf, +pos_b, LN -> query_embed (f32, rows < validRows)
// ---------------------------------------------------------------------------
template <int MODE>
__global__ __launch_bounds__(64) void k_gemm(
    const float* __restrict__ Af32, const _Float16* __restrict__ Af16,
    const _Float16* __restrict__ Bp, const float* __restrict__ bias,
    const float* __restrict__ gamma, const float* __restrict__ beta,
    _Float16* __restrict__ Hout, float* __restrict__ outA, float* __restrict__ scores,
    const float* __restrict__ wSmall, const float* __restrict__ bSmall,
    const unsigned char* __restrict__ mask, const float* __restrict__ vwh,
    float* __restrict__ cu_ws, int validRows) {
    __shared__ _Float16 Ash[2][16 * 256];
    __shared__ float Csh[2][16 * 256];
    const int w = threadIdx.x >> 5;
    const int lane = threadIdx.x & 31;
    const long tile = (long)blockIdx.x * 2 + w;
    const long R = tile * 16;

    // ---- Phase 1: stage A tile into LDS (f16) ----
    if (MODE == 0) {
        for (int i = 0; i < 16; ++i) {
            long rg = R + i;
            int b = (int)(rg / SS);
            int s = (int)(rg % SS);
            int x = s % WWID, y = s / WWID;
            float vw = vwh[b], vh = vwh[8 + b];
            float px = (x + 0.5f) / vw, py = (y + 0.5f) / vh;
            bool ok = (!mask[rg]) && (px > 0.01f) && (px < 0.99f) && (py > 0.01f) && (py < 0.99f);
            float okf = ok ? 1.0f : 0.0f;
            const float4* src = reinterpret_cast<const float4*>(Af32 + rg * 256);
            float4 v0 = src[lane];
            float4 v1 = src[lane + 32];
            _Float16* d = &Ash[w][i * 256];
            d[lane * 4 + 0] = (_Float16)(v0.x * okf);
            d[lane * 4 + 1] = (_Float16)(v0.y * okf);
            d[lane * 4 + 2] = (_Float16)(v0.z * okf);
            d[lane * 4 + 3] = (_Float16)(v0.w * okf);
            d[128 + lane * 4 + 0] = (_Float16)(v1.x * okf);
            d[128 + lane * 4 + 1] = (_Float16)(v1.y * okf);
            d[128 + lane * 4 + 2] = (_Float16)(v1.z * okf);
            d[128 + lane * 4 + 3] = (_Float16)(v1.w * okf);
        }
    } else {
        for (int i = 0; i < 16; ++i) {
            long rg = R + i;
            const uint4* src = reinterpret_cast<const uint4*>(Af16 + rg * 256);
            reinterpret_cast<uint4*>(&Ash[w][i * 256])[lane] = src[lane];
        }
    }
    __syncthreads();

    // ---- Phase 2: WMMA ----
    wave_gemm(Ash[w], Bp, Csh[w], lane);
    __syncthreads();

    // ---- Phase 3: epilogue (2 lanes per row, 128 cols each) ----
    const int r = lane >> 1;
    const int hf = lane & 1;
    const int c0 = hf * 128;
    long rg = R + r;
    float* Crow = &Csh[w][r * 256];

    if (MODE == 0 || MODE == 3) {
        float s1 = 0.f;
        for (int c = 0; c < 128; ++c) {
            float v = Crow[c0 + c] + bias[c0 + c];
            Crow[c0 + c] = v;
            s1 += v;
        }
        s1 += __shfl_xor(s1, 1, 32);
        float mean = s1 * (1.0f / 256.0f);
        float s2 = 0.f;
        for (int c = 0; c < 128; ++c) {
            float d = Crow[c0 + c] - mean;
            s2 += d * d;
        }
        s2 += __shfl_xor(s2, 1, 32);
        float inv = 1.0f / sqrtf(s2 * (1.0f / 256.0f) + 1e-5f);
        if (MODE == 0) {
            float d0 = 0.f, d1 = 0.f;
            for (int c = 0; c < 128; ++c) {
                int col = c0 + c;
                float h = (Crow[col] - mean) * inv * gamma[col] + beta[col];
                Hout[rg * 256 + col] = (_Float16)h;
                d0 += h * wSmall[col * 2 + 0];
                d1 += h * wSmall[col * 2 + 1];
            }
            d0 += __shfl_xor(d0, 1, 32);
            d1 += __shfl_xor(d1, 1, 32);
            if (hf == 0) {
                float e0 = d0 + bSmall[0], e1 = d1 + bSmall[1];
                outA[rg * 2 + 0] = e0;
                outA[rg * 2 + 1] = e1;
                scores[rg] = 1.0f / (1.0f + expf(e0 - e1));  // softmax[...,1]
            }
        } else {  // MODE 3: query_embed
            if (rg < validRows) {
                for (int c = 0; c < 128; ++c) {
                    int col = c0 + c;
                    outA[rg * 256 + col] = (Crow[col] - mean) * inv * gamma[col] + beta[col];
                }
            }
        }
    } else if (MODE == 1) {
        for (int c = 0; c < 128; ++c) {
            int col = c0 + c;
            float h = Crow[col] + bias[col];
            Hout[rg * 256 + col] = (_Float16)(h > 0.f ? h : 0.f);
        }
    } else {  // MODE 2: coord head
        float d0 = 0.f, d1 = 0.f;
        for (int c = 0; c < 128; ++c) {
            int col = c0 + c;
            float h = Crow[col] + bias[col];
            h = h > 0.f ? h : 0.f;
            d0 += h * wSmall[col * 2 + 0];
            d1 += h * wSmall[col * 2 + 1];
        }
        d0 += __shfl_xor(d0, 1, 32);
        d1 += __shfl_xor(d1, 1, 32);
        if (hf == 0) {
            int b = (int)(rg / SS);
            int s = (int)(rg % SS);
            int x = s % WWID, y = s / WWID;
            float vw = vwh[b], vh = vwh[8 + b];
            float px = (x + 0.5f) / vw, py = (y + 0.5f) / vh;
            bool ok = (!mask[rg]) && (px > 0.01f) && (px < 0.99f) && (py > 0.01f) && (py < 0.99f);
            float pu0 = ok ? logf(px / (1.0f - px)) : __builtin_inff();
            float pu1 = ok ? logf(py / (1.0f - py)) : __builtin_inff();
            float cu0 = d0 + bSmall[0] + pu0;
            float cu1 = d1 + bSmall[1] + pu1;
            cu_ws[rg * 2 + 0] = cu0;
            cu_ws[rg * 2 + 1] = cu1;
            // flipped sigmoid coords: out[...,0]=sig(cu_y), out[...,1]=sig(cu_x)
            outA[rg * 2 + 0] = 1.0f / (1.0f + expf(-cu1));
            outA[rg * 2 + 1] = 1.0f / (1.0f + expf(-cu0));
        }
    }
}

// ---------------------------------------------------------------------------
// K5: exact top-k (score desc, lower index first) via bit-threshold + bitonic sort
// ---------------------------------------------------------------------------
__global__ __launch_bounds__(1024) void k_topk(const float* __restrict__ scores,
                                               const float* __restrict__ cu_ws,
                                               float* __restrict__ tc) {
    __shared__ unsigned long long keys[4096];
    __shared__ int cntS;
    int b = blockIdx.x, tid = threadIdx.x;
    const float* sc = scores + (size_t)b * SS;
    unsigned lo = 0u, hi = 0x7F800000u;  // scores in (0,1): positive, bit-monotonic
    for (int it = 0; it < 31; ++it) {
        if (hi - lo <= 1u) break;
        unsigned mid = lo + ((hi - lo) >> 1);
        if (tid == 0) cntS = 0;
        __syncthreads();
        int local = 0;
        for (int i = tid; i < SS; i += 1024)
            local += (__float_as_uint(sc[i]) >= mid) ? 1 : 0;
        atomicAdd(&cntS, local);
        __syncthreads();
        int c = cntS;
        __syncthreads();
        if (c >= NQ) lo = mid; else hi = mid;
    }
    unsigned bstar = lo;  // nq-th largest score bits; count(>=bstar) >= NQ
    if (tid == 0) cntS = 0;
    __syncthreads();
    for (int i = tid; i < SS; i += 1024) {
        unsigned bits = __float_as_uint(sc[i]);
        if (bits >= bstar) {
            int slot = atomicAdd(&cntS, 1);
            if (slot < 4096)
                keys[slot] = ((unsigned long long)bits << 32) | (unsigned)(~(unsigned)i);
        }
    }
    __syncthreads();
    int total = cntS;
    if (total > 4096) total = 4096;
    for (int i = total + tid; i < 4096; i += 1024) keys[i] = 0ull;
    __syncthreads();
    // bitonic sort, descending (ties -> larger ~index first == smaller index first)
    for (unsigned k = 2; k <= 4096u; k <<= 1) {
        for (unsigned j = k >> 1; j > 0; j >>= 1) {
            for (unsigned i = tid; i < 4096u; i += 1024u) {
                unsigned ixj = i ^ j;
                if (ixj > i) {
                    unsigned long long a = keys[i], c = keys[ixj];
                    bool desc = ((i & k) == 0u);
                    if (desc ? (a < c) : (a > c)) { keys[i] = c; keys[ixj] = a; }
                }
            }
            __syncthreads();
        }
    }
    for (int i = tid; i < NQ; i += 1024) {
        unsigned long long kv = keys[i];
        unsigned idx = ~((unsigned)(kv & 0xFFFFFFFFull));
        tc[((size_t)b * NQ + i) * 2 + 0] = cu_ws[((size_t)b * SS + idx) * 2 + 0];
        tc[((size_t)b * NQ + i) * 2 + 1] = cu_ws[((size_t)b * SS + idx) * 2 + 1];
    }
}

// ---------------------------------------------------------------------------
// K6: per-query: pos-embed (f16 -> posbuf), grid-sample query_feats, div gather
// ---------------------------------------------------------------------------
__global__ void k_query(const float* __restrict__ tc, const float* __restrict__ bfeat,
                        const float* __restrict__ divp, _Float16* __restrict__ posb,
                        float* __restrict__ outQF, float* __restrict__ outDiv) {
    int row = blockIdx.x, t = threadIdx.x;
    if (row >= RQ) {  // zero pad rows for the pos GEMM
        posb[(size_t)row * 256 + t] = (_Float16)0.0f;
        return;
    }
    int b = row / NQ;
    float cu0 = tc[(size_t)row * 2 + 0];
    float cu1 = tc[(size_t)row * 2 + 1];
    float rx = 1.0f / (1.0f + expf(-cu0));
    float ry = 1.0f / (1.0f + expf(-cu1));
    // pos embed: p flipped -> d=0 uses y, d=1 uses x; out[d*128 + 2j + {0:sin,1:cos}]
    int d = t >> 7, r = t & 127, jj = r >> 1;
    float pp = (d == 0 ? ry : rx) * 6.283185307179586f;
    float freq = powf(10000.0f, (float)jj * (1.0f / 64.0f));
    float val = pp / freq;
    posb[(size_t)row * 256 + t] = (_Float16)(((r & 1) == 0) ? sinf(val) : cosf(val));
    // bilinear grid-sample, zero padding, align_corners=False; channel = t
    float ix = rx * 128.0f - 0.5f;
    float iy = ry * 96.0f - 0.5f;
    float x0 = floorf(ix), y0 = floorf(iy);
    float x1 = x0 + 1.0f, y1 = y0 + 1.0f;
    float wa = (x1 - ix) * (y1 - iy), wb2 = (ix - x0) * (y1 - iy);
    float wc = (x1 - ix) * (iy - y0), wd = (ix - x0) * (iy - y0);
    const float* img = bfeat + ((size_t)b * 256 + t) * 12288;
    auto tap = [&](float xf, float yf) -> float {
        if (xf < 0.f || xf > 127.f || yf < 0.f || yf > 95.f) return 0.f;
        return img[(int)yf * 128 + (int)xf];
    };
    outQF[(size_t)row * 256 + t] =
        tap(x0, y0) * wa + tap(x1, y0) * wb2 + tap(x0, y1) * wc + tap(x1, y1) * wd;
    if (t == 0) {
        float px = rintf(rx * 128.0f), py = rintf(ry * 96.0f);  // round half-to-even
        int idx = (int)py * 128 + (int)px;
        idx = idx < 0 ? 0 : (idx > 12287 ? 12287 : idx);
        outDiv[row] = divp[(size_t)b * 12288 + idx];
    }
}

// ---------------------------------------------------------------------------
extern "C" void kernel_launch(void* const* d_in, const int* in_sizes, int n_in,
                              void* d_out, int out_size, void* d_ws, size_t ws_size,
                              hipStream_t stream) {
    (void)in_sizes; (void)n_in; (void)out_size; (void)ws_size;
    const float* memory = (const float*)d_in[0];
    const unsigned char* mask = (const unsigned char*)d_in[1];
    const float* bfeat = (const float*)d_in[2];
    const float* divp = (const float*)d_in[3];
    const float* enc_w = (const float*)d_in[4];
    const float* enc_b = (const float*)d_in[5];
    const float* enc_ng = (const float*)d_in[6];
    const float* enc_nb = (const float*)d_in[7];
    const float* cls_w = (const float*)d_in[8];
    const float* cls_b = (const float*)d_in[9];
    const float* w1 = (const float*)d_in[10];
    const float* b1 = (const float*)d_in[11];
    const float* w2 = (const float*)d_in[12];
    const float* b2 = (const float*)d_in[13];
    const float* w3 = (const float*)d_in[14];
    const float* b3 = (const float*)d_in[15];
    const float* pos_w = (const float*)d_in[16];
    const float* pos_b = (const float*)d_in[17];
    const float* pos_ng = (const float*)d_in[18];
    const float* pos_nb = (const float*)d_in[19];

    char* ws = (char*)d_ws;
    float* vwh      = (float*)(ws + 0);              // 16 f32
    _Float16* Penc  = (_Float16*)(ws + 256);         // 131072 B each packed weight
    _Float16* Pw1   = (_Float16*)(ws + 131328);
    _Float16* Pw2   = (_Float16*)(ws + 262400);
    _Float16* Ppos  = (_Float16*)(ws + 393472);
    float* scores   = (float*)(ws + 524544);         // 98304 f32
    float* cu_ws    = (float*)(ws + 917760);         // 98304*2 f32
    float* tc       = (float*)(ws + 1704192);        // 19656*2 f32
    _Float16* H0    = (_Float16*)(ws + 1861632);     // 98304*256 f16
    _Float16* H1    = (_Float16*)(ws + 52193280);    // 98304*256 f16
    _Float16* posb  = (_Float16*)(ws + 102524928);   // 19680*256 f16  (end ~112.6 MB)

    float* out = (float*)d_out;
    float* outCls   = out + 0;         // [8,12288,2]
    float* outCoord = out + 196608;    // [8,12288,2]
    float* outQE    = out + 393216;    // [8,2457,256]
    float* outQF    = out + 5425152;   // [8,2457,256]
    float* outDiv   = out + 10457088;  // [8,2457]

    k_valid<<<8, 128, 0, stream>>>(mask, vwh);
    k_pack<<<1024, 256, 0, stream>>>(enc_w, w1, w2, pos_w, Penc, Pw1, Pw2, Ppos);
    k_gemm<0><<<3072, 64, 0, stream>>>(memory, nullptr, Penc, enc_b, enc_ng, enc_nb,
                                       H0, outCls, scores, cls_w, cls_b, mask, vwh,
                                       nullptr, ROWS);
    k_gemm<1><<<3072, 64, 0, stream>>>(nullptr, H0, Pw1, b1, nullptr, nullptr,
                                       H1, nullptr, nullptr, nullptr, nullptr, nullptr,
                                       nullptr, nullptr, ROWS);
    k_gemm<2><<<3072, 64, 0, stream>>>(nullptr, H1, Pw2, b2, nullptr, nullptr,
                                       nullptr, outCoord, nullptr, w3, b3, mask, vwh,
                                       cu_ws, ROWS);
    k_topk<<<8, 1024, 0, stream>>>(scores, cu_ws, tc);
    k_query<<<RQP, 256, 0, stream>>>(tc, bfeat, divp, posb, outQF, outDiv);
    k_gemm<3><<<615, 64, 0, stream>>>(nullptr, posb, Ppos, pos_b, pos_ng, pos_nb,
                                      nullptr, outQE, nullptr, nullptr, nullptr, nullptr,
                                      nullptr, nullptr, RQ);
}
